// EarthAttention3D_9698036154398
// MI455X (gfx1250) — matrix operand order repaired and checked by reading, not compiled, not run on hardware
//
#include <hip/hip_runtime.h>
#include <hip/hip_bf16.h>

typedef __attribute__((ext_vector_type(16))) _Float16 v16h;
typedef __attribute__((ext_vector_type(8)))  float    v8f;

#define HEADS 6
#define DH    32
#define ZHW   144
#define NWIN  640
#define DIM   192
#define QKVN  (3*DIM)            // 576
#define NTOK  (NWIN*ZHW)         // 92160

// ---------------------------------------------------------------------------
// CDNA5 async copy: global memory -> LDS, no VGPR data path (ASYNCcnt).
// dst is an LDS byte offset (low 32 bits of a generic LDS pointer), 16B aligned.
// ---------------------------------------------------------------------------
__device__ __forceinline__ void async_ld_b128(uint32_t lds_off, const void* gaddr) {
  asm volatile("global_load_async_to_lds_b128 %0, %1, off"
               :: "v"(lds_off), "v"(gaddr) : "memory");
}
__device__ __forceinline__ void wait_async0() {
  asm volatile("s_wait_asynccnt 0x0" ::: "memory");
}
__device__ __forceinline__ uint32_t lds_off32(const void* p) {
  return (uint32_t)(uintptr_t)p;   // generic LDS addr: low 32 bits are the LDS offset
}

// ---------------------------------------------------------------------------
// WMMA helper (CDNA5 wave32, V_WMMA_F32_16X16X32_F16)
// ---------------------------------------------------------------------------
__device__ __forceinline__ v8f wmma16(v16h a, v16h b, v8f c) {
  return __builtin_amdgcn_wmma_f32_16x16x32_f16(false, a, false, b, (short)0, c,
                                                false, false);
}

// A fragment: 16x32 f16, row-major in LDS, stride in halves. 2x ds_load_b128.
__device__ __forceinline__ v16h frag_a(const _Float16* base, int stride, int lane) {
  const int m = lane & 15, g = (lane >> 4) & 1;
  const _Float16* row = base + m * stride;
  v16h a;
#pragma unroll
  for (int i = 0; i < 16; ++i) a[i] = row[16*g + i];
  return a;
}

// B fragment: 32x16, B[k][n] = colmaj[n][k] (stride in halves). 2x ds_load_b128.
__device__ __forceinline__ v16h frag_bT(const _Float16* colbase, int stride, int lane) {
  const int n = lane & 15, g = (lane >> 4) & 1;
  const _Float16* row = colbase + n * stride + 16*g;
  v16h b;
#pragma unroll
  for (int i = 0; i < 16; ++i) b[i] = row[i];
  return b;
}

// ---------------------------------------------------------------------------
// Weight convert + transpose: in (K x N f32, row-major) -> out (N x K f16)
// ---------------------------------------------------------------------------
__global__ void k_wcvt(const float* __restrict__ in, _Float16* __restrict__ out,
                       int K, int N) {
  int i = blockIdx.x * 256 + threadIdx.x;
  if (i < K * N) {
    int k = i / N, n = i % N;
    out[(size_t)n * K + k] = (_Float16)in[i];
  }
}

// ---------------------------------------------------------------------------
// Fused earth bias: fused[tw*6+h][m*144+n] = bias_table[pos_index[m*144+n], tw, h]
// ---------------------------------------------------------------------------
__global__ __launch_bounds__(256) void k_bias(const float* __restrict__ bias_table,
    const int* __restrict__ pos_index, float* __restrict__ fused) {
  const int twh = blockIdx.x;                 // 0..383
  const int tw = twh / HEADS, h = twh % HEADS;
  float* dst = fused + (size_t)twh * (ZHW*ZHW);
  for (int e = threadIdx.x; e < ZHW*ZHW; e += 256) {
    int idx = pos_index[e];
    dst[e] = bias_table[(size_t)idx * (64*HEADS) + tw*HEADS + h];
  }
}

// ---------------------------------------------------------------------------
// QKV GEMM: (92160 x 192) x (192 x 576) + b1 -> scatter q/k f16 [win][h][tok][d],
// v f16 transposed [win][h][d][tok]. Block tile 256x64, 8 waves of 32x64.
// ---------------------------------------------------------------------------
__global__ __launch_bounds__(256) void k_qkv(const float* __restrict__ x,
    const _Float16* __restrict__ w1t, const float* __restrict__ b1,
    _Float16* __restrict__ qh, _Float16* __restrict__ kh, _Float16* __restrict__ vh) {
  __shared__ __align__(16) _Float16 As[256*32];   // 16 KB
  __shared__ __align__(16) _Float16 Bt[64*32];    // 4 KB, col-major [n][k]
  const int tid = threadIdx.x, lane = tid & 31, wave = tid >> 5;
  const int M0 = blockIdx.x * 256, N0 = blockIdx.y * 64;
  const uint32_t Boff = lds_off32(Bt);
  v8f acc[2][4] = {};
  for (int k0 = 0; k0 < DIM; k0 += 32) {
    // B tile: async copy, 64 rows x 64B; one b128 per thread
    async_ld_b128(Boff + tid*16,
                  (const char*)w1t + ((size_t)(N0 + (tid >> 2)) * DIM + k0) * 2
                                   + (tid & 3) * 16);
    // A tile: one 32-float row per thread, f32 -> packed f16, 4x b128 LDS store
    {
      const float4* xr = (const float4*)(x + (size_t)(M0 + tid) * DIM + k0);
      union { _Float16 h[32]; uint4 u4[4]; } p;
#pragma unroll
      for (int q = 0; q < 8; ++q) {
        float4 f = xr[q];
        p.h[4*q + 0] = (_Float16)f.x; p.h[4*q + 1] = (_Float16)f.y;
        p.h[4*q + 2] = (_Float16)f.z; p.h[4*q + 3] = (_Float16)f.w;
      }
      uint4* dst = (uint4*)(As + tid * 32);
#pragma unroll
      for (int q = 0; q < 4; ++q) dst[q] = p.u4[q];
    }
    if (k0 + 32 < DIM)
      __builtin_prefetch(&x[(size_t)(M0 + tid) * DIM + k0 + 32], 0, 1);
    wait_async0();
    __syncthreads();
    v16h a0 = frag_a(As + (wave*32)      * 32, 32, lane);
    v16h a1 = frag_a(As + (wave*32 + 16) * 32, 32, lane);
#pragma unroll
    for (int j = 0; j < 4; ++j) {
      v16h bf = frag_bT(Bt + (16*j) * 32, 32, lane);
      acc[0][j] = wmma16(a0, bf, acc[0][j]);
      acc[1][j] = wmma16(a1, bf, acc[1][j]);
    }
    __syncthreads();
  }
  const int g = (lane >> 4) & 1, nl = lane & 15;
#pragma unroll
  for (int i = 0; i < 2; ++i)
#pragma unroll
    for (int j = 0; j < 4; ++j)
#pragma unroll
      for (int r = 0; r < 8; ++r) {
        int m = M0 + wave*32 + 16*i + 8*g + r;
        int n = N0 + 16*j + nl;
        float val = acc[i][j][r] + b1[n];
        int win = m / ZHW, tok = m % ZHW;
        int sel = n / DIM, head = (n % DIM) >> 5, d = n & 31;
        size_t wb = (size_t)(win*HEADS + head);
        if (sel == 0)      qh[(wb*ZHW + tok)*DH + d] = (_Float16)val;
        else if (sel == 1) kh[(wb*ZHW + tok)*DH + d] = (_Float16)val;
        else               vh[(wb*DH + d)*ZHW + tok] = (_Float16)val;  // transposed
      }
}

// ---------------------------------------------------------------------------
// Fused attention, one block per (win, head). 9 waves, wave i owns rows
// [16i,16i+16). q/k/v staged via async LDS loads; fragments are 2x ds_load_b128.
// ---------------------------------------------------------------------------
__global__ __launch_bounds__(288) void k_attn(const _Float16* __restrict__ qh,
    const _Float16* __restrict__ kh, const _Float16* __restrict__ vh,
    const float* __restrict__ mask, const float* __restrict__ fbias,
    _Float16* __restrict__ oh) {
  __shared__ __align__(16) _Float16 qs[ZHW*DH];   // [tok][d]
  __shared__ __align__(16) _Float16 ks[ZHW*DH];   // [tok][d] == colmaj for S B-frag
  __shared__ __align__(16) _Float16 vst[DH*160];  // [d][tok], tokens 144..159 zero
  __shared__ __align__(16) _Float16 ps[9*16*32];  // per-wave P chunk
  const int wh = blockIdx.x;
  const int win = wh / HEADS, head = wh % HEADS;
  const int tw = win & 63;
  const int tid = threadIdx.x, lane = tid & 31, wave = tid >> 5;
  const int g = (lane >> 4) & 1, nl = lane & 15;
  const size_t base = (size_t)wh * (ZHW*DH);
  {
    const char* qg = (const char*)(qh + base);
    const char* kg = (const char*)(kh + base);
    const char* vg = (const char*)(vh + base);    // [d][tok]: rows of 288B
    const uint32_t qoff = lds_off32(qs), koff = lds_off32(ks), voff = lds_off32(vst);
    // q, k: 9216B each = 576 b128; 2 per thread
    for (int e = tid; e < 576; e += 288) {
      async_ld_b128(qoff + e*16, qg + e*16);
      async_ld_b128(koff + e*16, kg + e*16);
    }
    // v: 32 d-rows x 18 b128 valid, LDS row pitch 20 b128 (320B)
    for (int e = tid; e < 576; e += 288) {
      int d = e / 18, c = e - d*18;
      async_ld_b128(voff + (d*20 + c)*16, vg + e*16);
    }
    // zero pad tokens 144..159 (2 b128 per d-row)
    if (tid < 64) {
      uint4 z = {0u, 0u, 0u, 0u};
      *(uint4*)((char*)vst + ((tid >> 1)*20 + 18 + (tid & 1))*16) = z;
    }
    wait_async0();
  }
  __syncthreads();

  v16h aq = frag_a(qs + wave*16*DH, DH, lane);
  v8f s[9];
#pragma unroll
  for (int t = 0; t < 9; ++t) {
    v16h bk = frag_bT(ks + t*16*DH, DH, lane);    // B[d][tok] = ks[tok][d]
    v8f z = {};
    s[t] = wmma16(aq, bk, z);
  }
  const float invs = 0.17677669529663687f;        // 1/sqrt(32)
  const float* fb = fbias + (size_t)(tw*HEADS + head) * (ZHW*ZHW);
  const float* mk = mask + (size_t)win * (ZHW*ZHW);
#pragma unroll
  for (int t = 0; t < 9; ++t)
#pragma unroll
    for (int r = 0; r < 8; ++r) {
      int mn = (wave*16 + 8*g + r) * ZHW + t*16 + nl;
      s[t][r] = s[t][r] * invs + fb[mn] + mk[mn];
    }
  // row softmax over 144 cols: local 9-tile reduce + 16-lane shuffle reduce
#pragma unroll
  for (int r = 0; r < 8; ++r) {
    float mx = s[0][r];
#pragma unroll
    for (int t = 1; t < 9; ++t) mx = fmaxf(mx, s[t][r]);
    mx = fmaxf(mx, __shfl_xor(mx, 1, 32));
    mx = fmaxf(mx, __shfl_xor(mx, 2, 32));
    mx = fmaxf(mx, __shfl_xor(mx, 4, 32));
    mx = fmaxf(mx, __shfl_xor(mx, 8, 32));
    float sum = 0.f;
#pragma unroll
    for (int t = 0; t < 9; ++t) { float p = __expf(s[t][r] - mx); s[t][r] = p; sum += p; }
    sum += __shfl_xor(sum, 1, 32);
    sum += __shfl_xor(sum, 2, 32);
    sum += __shfl_xor(sum, 4, 32);
    sum += __shfl_xor(sum, 8, 32);
    float rs = 1.f / sum;
#pragma unroll
    for (int t = 0; t < 9; ++t) s[t][r] *= rs;
  }
  // O = P * V, 5 token-chunks of 32 (chunk 4 zero-padded)
  _Float16* myps = ps + wave*16*32;               // wave-private
  v8f o0 = {}, o1 = {};
#pragma unroll
  for (int c = 0; c < 5; ++c) {
#pragma unroll
    for (int cc = 0; cc < 2; ++cc) {
      const int t = 2*c + cc;
      const int ts = (t < 9) ? t : 0;
#pragma unroll
      for (int r = 0; r < 8; ++r) {
        float pv = (t < 9) ? s[ts][r] : 0.f;
        myps[(8*g + r)*32 + 16*cc + nl] = (_Float16)pv;
      }
    }
    v16h ap  = frag_a(myps, 32, lane);
    v16h bv0 = frag_bT(vst +          32*c, 160, lane);   // d-tile 0
    v16h bv1 = frag_bT(vst + 16*160 + 32*c, 160, lane);   // d-tile 1
    o0 = wmma16(ap, bv0, o0);
    o1 = wmma16(ap, bv1, o1);
  }
  // torch-faithful swapaxes(1,2): flat = ((head*NWIN+win)*ZHW+tok)*DH+d
#pragma unroll
  for (int r = 0; r < 8; ++r) {
    int tok = wave*16 + 8*g + r;
    size_t fo = (((size_t)(head*NWIN + win)) * ZHW + tok) * DH;
    oh[fo + nl]      = (_Float16)o0[r];
    oh[fo + 16 + nl] = (_Float16)o1[r];
  }
}

// ---------------------------------------------------------------------------
// Output projection: (92160 x 192) x (192 x 192) + b2 -> f32 d_out.
// Block tile 256x64, 8 waves of 32x64; A and B tiles staged with async LDS loads.
// ---------------------------------------------------------------------------
__global__ __launch_bounds__(256) void k_proj(const _Float16* __restrict__ oh,
    const _Float16* __restrict__ w2t, const float* __restrict__ b2,
    float* __restrict__ out) {
  __shared__ __align__(16) _Float16 As[256*32];
  __shared__ __align__(16) _Float16 Bt[64*32];
  const int tid = threadIdx.x, lane = tid & 31, wave = tid >> 5;
  const int M0 = blockIdx.x * 256, N0 = blockIdx.y * 64;
  const uint32_t Aoff = lds_off32(As), Boff = lds_off32(Bt);
  v8f acc[2][4] = {};
  for (int k0 = 0; k0 < DIM; k0 += 32) {
    // A tile: 256 rows x 64B, 4 b128 per thread (one row each)
    const char* ar = (const char*)oh + ((size_t)(M0 + tid) * DIM + k0) * 2;
#pragma unroll
    for (int q = 0; q < 4; ++q)
      async_ld_b128(Aoff + tid*64 + q*16, ar + q*16);
    // B tile: 64 rows x 64B, 1 b128 per thread
    async_ld_b128(Boff + tid*16,
                  (const char*)w2t + ((size_t)(N0 + (tid >> 2)) * DIM + k0) * 2
                                   + (tid & 3) * 16);
    wait_async0();
    __syncthreads();
    v16h a0 = frag_a(As + (wave*32)      * 32, 32, lane);
    v16h a1 = frag_a(As + (wave*32 + 16) * 32, 32, lane);
#pragma unroll
    for (int j = 0; j < 4; ++j) {
      v16h bf = frag_bT(Bt + (16*j) * 32, 32, lane);
      acc[0][j] = wmma16(a0, bf, acc[0][j]);
      acc[1][j] = wmma16(a1, bf, acc[1][j]);
    }
    __syncthreads();
  }
  const int g = (lane >> 4) & 1, nl = lane & 15;
#pragma unroll
  for (int i = 0; i < 2; ++i)
#pragma unroll
    for (int j = 0; j < 4; ++j)
#pragma unroll
      for (int r = 0; r < 8; ++r) {
        int m = M0 + wave*32 + 16*i + 8*g + r;
        int n = N0 + 16*j + nl;
        out[(size_t)m * DIM + n] = acc[i][j][r] + b2[n];
      }
}

// ---------------------------------------------------------------------------
extern "C" void kernel_launch(void* const* d_in, const int* in_sizes, int n_in,
                              void* d_out, int out_size, void* d_ws, size_t ws_size,
                              hipStream_t stream) {
  const float* x          = (const float*)d_in[0];
  const float* mask       = (const float*)d_in[1];
  const float* w1         = (const float*)d_in[2];
  const float* b1         = (const float*)d_in[3];
  const float* w2         = (const float*)d_in[4];
  const float* b2         = (const float*)d_in[5];
  const float* bias_table = (const float*)d_in[6];
  const int*   pos_index  = (const int*)d_in[7];
  float* out = (float*)d_out;

  const size_t nQKV = (size_t)NWIN * ZHW * DIM;   // 17,694,720 elements
  _Float16* qh  = (_Float16*)d_ws;
  _Float16* kh  = qh + nQKV;
  _Float16* vh  = kh + nQKV;
  _Float16* oh  = vh + nQKV;
  _Float16* w1t = oh + nQKV;
  _Float16* w2t = w1t + (size_t)DIM * QKVN;
  float*    fbias = (float*)(w2t + (size_t)DIM * DIM);  // 64*6*144*144 f32

  k_wcvt<<<dim3((DIM*QKVN + 255) / 256), dim3(256), 0, stream>>>(w1, w1t, DIM, QKVN);
  k_wcvt<<<dim3((DIM*DIM  + 255) / 256), dim3(256), 0, stream>>>(w2, w2t, DIM, DIM);
  k_bias<<<dim3(64*HEADS), dim3(256), 0, stream>>>(bias_table, pos_index, fbias);
  k_qkv<<<dim3(NTOK/256, QKVN/64), dim3(256), 0, stream>>>(x, w1t, b1, qh, kh, vh);
  k_attn<<<dim3(NWIN*HEADS), dim3(288), 0, stream>>>(qh, kh, vh, mask, fbias, oh);
  k_proj<<<dim3(NTOK/256, DIM/64), dim3(256), 0, stream>>>(oh, w2t, b2, out);
}
